// TransLayer_43593918054939
// MI455X (gfx1250) — matrix operand order repaired and checked by reading, not compiled, not run on hardware
//
#include <hip/hip_runtime.h>
#include <hip/hip_bf16.h>

// ---------------- problem constants ----------------
#define B_SZ   4
#define N_SEQ  8192
#define D_MODEL 512
#define HEADS  8
#define DH     64
#define INNER  512
#define M_LM   256
#define L_GRP  32
#define BH     (B_SZ*HEADS)          // 32
#define ROWS   ((long long)B_SZ*N_SEQ) // 32768
#define PINV_ITERS 6
#define KER    33
#define LDSK   40                     // 32 + 8 bf16 pad (row = 80B, 16B aligned)

typedef __bf16 bf16;
typedef __attribute__((ext_vector_type(16))) __bf16 v16bf;
typedef __attribute__((ext_vector_type(8)))  float  v8f;

__device__ __forceinline__ bf16 f2bf(float f) {
  unsigned u = __float_as_uint(f);
  unsigned r = u + 0x7fffu + ((u >> 16) & 1u);   // round-to-nearest-even
  unsigned short hs = (unsigned short)(r >> 16);
  bf16 out; __builtin_memcpy(&out, &hs, 2); return out;
}
__device__ __forceinline__ float bf2f(bf16 b) {
  unsigned short hs; __builtin_memcpy(&hs, &b, 2);
  unsigned u = ((unsigned)hs) << 16;
  float f; __builtin_memcpy(&f, &u, 4); return f;
}

// ---------------- generic batched WMMA GEMM (bf16 in, fp32 accum) ----------
// C[M,N] = alpha * A[M,K] x (transB ? B[N,K]^T : B[K,N])  (+ bias + resid)
// Block tile 128x128, BK=32, 8 wave32 waves -> each wave 32x64 (2x4 WMMA tiles)
// Double-buffered LDS tiles; A (and transposed-B) staged with
// GLOBAL_LOAD_ASYNC_TO_LDS_B128 (ASYNCcnt pipeline), overlapping DMA of tile
// k+1 with the 8 WMMAs of tile k.
__global__ __launch_bounds__(256)
void wmma_gemm_bf16(const bf16* __restrict__ A, long long strideA, int lda,
                    const bf16* __restrict__ Bm, long long strideB, int ldb, int transB,
                    float* __restrict__ C, bf16* __restrict__ Dbf,
                    long long strideC, int ldc,
                    int M, int N, int K, float alpha,
                    const float* __restrict__ bias,
                    const float* __restrict__ resid)
{
  __shared__ __align__(16) bf16 As[2][128][LDSK];
  __shared__ __align__(16) bf16 Bs[2][128][LDSK];

  const int tid  = threadIdx.x;
  const int lane = tid & 31;
  const int wave = tid >> 5;
  const int wr   = wave >> 1;      // 0..3 : 32-row stripes
  const int wc   = wave & 1;       // 0..1 : 64-col stripes
  const int hf   = lane >> 4;      // lane half (ISA fragment layouts)
  const int l16  = lane & 15;

  const int bRow = blockIdx.y * 128;
  const int bCol = blockIdx.x * 128;
  const int bat  = blockIdx.z;

  const bf16* Ab = A  + (long long)bat * strideA;
  const bf16* Bb = Bm + (long long)bat * strideB;

  v8f acc[2][4];
  #pragma unroll
  for (int i = 0; i < 2; ++i)
    #pragma unroll
    for (int j = 0; j < 4; ++j)
      #pragma unroll
      for (int e = 0; e < 8; ++e) acc[i][j][e] = 0.f;

  const int nk   = K >> 5;          // K is always a multiple of 32 here
  const int arow = tid >> 1;        // 0..127
  const int aseg = tid & 1;         // two 16-elem (32B) segments
  const int bk   = tid >> 3;        // 0..31 (non-transposed B loader)
  const int bns  = tid & 7;         // 0..7

  // ---- stage tile kt into buffer b ----
  auto stage = [&](int kt, int b) {
    const int k0 = kt << 5;
    { // A tile: async DMA, 2 x b128 per lane (INST_OFFSET hits LDS + global)
      const int gr = bRow + arow;
      if (gr < M) {   // always true in this app (M % 128 == 0)
        const unsigned long long ga =
            (unsigned long long)(Ab + (long long)gr * lda + (k0 + aseg * 16));
        const unsigned lds0 =
            (unsigned)(unsigned long long)&As[b][arow][aseg * 16];
        asm volatile("global_load_async_to_lds_b128 %0, %1, off"
                     :: "v"(lds0), "v"(ga) : "memory");
        asm volatile("global_load_async_to_lds_b128 %0, %1, off offset:16"
                     :: "v"(lds0), "v"(ga) : "memory");
      } else {
        uint4 z = make_uint4(0,0,0,0);
        *(uint4*)&As[b][arow][aseg * 16]     = z;
        *(uint4*)&As[b][arow][aseg * 16 + 8] = z;
      }
    }
    if (transB) { // B' is [N,K] row-major: Bs[n][k] direct async copy
      const int gn = bCol + arow;
      if (gn < N) { // always true for transB calls here (N % 128 == 0)
        const unsigned long long gb =
            (unsigned long long)(Bb + (long long)gn * ldb + (k0 + aseg * 16));
        const unsigned lds0 =
            (unsigned)(unsigned long long)&Bs[b][arow][aseg * 16];
        asm volatile("global_load_async_to_lds_b128 %0, %1, off"
                     :: "v"(lds0), "v"(gb) : "memory");
        asm volatile("global_load_async_to_lds_b128 %0, %1, off offset:16"
                     :: "v"(lds0), "v"(gb) : "memory");
      } else {
        uint4 z = make_uint4(0,0,0,0);
        *(uint4*)&Bs[b][arow][aseg * 16]     = z;
        *(uint4*)&Bs[b][arow][aseg * 16 + 8] = z;
      }
    } else {      // B is [K,N] row-major: transpose into Bs[n][k] (sync path)
      const int gk = k0 + bk;
      const int n0 = bCol + bns * 16;
      union { uint4 u[2]; bf16 e[16]; } tb;
      if (n0 + 15 < N) {
        const bf16* src = Bb + (long long)gk * ldb + n0;
        tb.u[0] = *(const uint4*)src; tb.u[1] = *(const uint4*)(src + 8);
        if (kt + 1 < nk) __builtin_prefetch(src + 32ll * ldb, 0, 1); // prefetch_b8
      } else {
        #pragma unroll
        for (int e = 0; e < 16; ++e)
          tb.e[e] = (n0 + e < N) ? Bb[(long long)gk * ldb + n0 + e] : f2bf(0.f);
      }
      #pragma unroll
      for (int e = 0; e < 16; ++e) Bs[b][bns * 16 + e][bk] = tb.e[e];
    }
  };

  stage(0, 0);   // prologue

  for (int kt = 0; kt < nk; ++kt) {
    const int cur = kt & 1;
    if (kt + 1 < nk) stage(kt + 1, cur ^ 1);

    // current tile's async ops are the oldest; async loads retire in order,
    // so waiting down to the next tile's in-flight count fences tile kt.
    if (kt + 1 < nk) {
      if (transB) asm volatile("s_wait_asynccnt 0x4" ::: "memory");
      else        asm volatile("s_wait_asynccnt 0x2" ::: "memory");
    } else {
      asm volatile("s_wait_asynccnt 0x0" ::: "memory");
    }
    __syncthreads();

    // ---- fragments per ISA layouts ----
    union Frag { v16bf v; uint4 u[2]; };
    Frag af[2], bfr[4];
    #pragma unroll
    for (int mi = 0; mi < 2; ++mi) {         // A 16x32: K = hf*8+j | 16+hf*8+j
      const int r = wr * 32 + mi * 16 + l16;
      af[mi].u[0] = *(const uint4*)&As[cur][r][hf * 8];
      af[mi].u[1] = *(const uint4*)&As[cur][r][16 + hf * 8];
    }
    #pragma unroll
    for (int ni = 0; ni < 4; ++ni) {         // B 32x16: K = hf*8+j | 16+hf*8+j for col n
      const int n = wc * 64 + ni * 16 + l16;
      bfr[ni].u[0] = *(const uint4*)&Bs[cur][n][hf * 8];
      bfr[ni].u[1] = *(const uint4*)&Bs[cur][n][16 + hf * 8];
    }
    #pragma unroll
    for (int mi = 0; mi < 2; ++mi)
      #pragma unroll
      for (int ni = 0; ni < 4; ++ni)
        acc[mi][ni] = __builtin_amdgcn_wmma_f32_16x16x32_bf16(
            false, af[mi].v, false, bfr[ni].v, (short)0, acc[mi][ni], false, false);
    __syncthreads();
  }

  // ---- epilogue: C VGPR v holds row = v + 8*hf, col = l16 ----
  float* Cb = C   ? C   + (long long)bat * strideC : nullptr;
  bf16*  Db = Dbf ? Dbf + (long long)bat * strideC : nullptr;
  const float* Rb = resid ? resid + (long long)bat * strideC : nullptr;
  #pragma unroll
  for (int mi = 0; mi < 2; ++mi)
    #pragma unroll
    for (int ni = 0; ni < 4; ++ni)
      #pragma unroll
      for (int vv = 0; vv < 8; ++vv) {
        const int row = bRow + wr * 32 + mi * 16 + hf * 8 + vv;
        const int col = bCol + wc * 64 + ni * 16 + l16;
        if (row < M && col < N) {
          float val = alpha * acc[mi][ni][vv];
          if (bias) val += bias[col];
          long long o = (long long)row * ldc + col;
          if (Rb) val += Rb[o];
          if (Cb) Cb[o] = val;
          if (Db) Db[o] = f2bf(val);
        }
      }
}

// ---------------- LayerNorm -> bf16 ----------------
__global__ __launch_bounds__(256)
void ln_kernel(const float* __restrict__ x, const float* __restrict__ w,
               const float* __restrict__ bv, bf16* __restrict__ hout)
{
  __shared__ float r1[256], r2[256];
  const long long row = blockIdx.x;
  const float* xr = x + row * D_MODEL;
  float s = 0.f, s2 = 0.f;
  for (int i = threadIdx.x; i < D_MODEL; i += 256) { float v = xr[i]; s += v; s2 += v * v; }
  r1[threadIdx.x] = s; r2[threadIdx.x] = s2; __syncthreads();
  for (int st = 128; st > 0; st >>= 1) {
    if (threadIdx.x < st) { r1[threadIdx.x] += r1[threadIdx.x + st]; r2[threadIdx.x] += r2[threadIdx.x + st]; }
    __syncthreads();
  }
  const float mu  = r1[0] * (1.f / D_MODEL);
  const float var = r2[0] * (1.f / D_MODEL) - mu * mu;
  const float inv = rsqrtf(var + 1e-5f);
  for (int i = threadIdx.x; i < D_MODEL; i += 256)
    hout[row * D_MODEL + i] = f2bf((xr[i] - mu) * inv * w[i] + bv[i]);
}

// ---------------- fp32 -> bf16 cast ----------------
__global__ void cast_kernel(const float* __restrict__ in, bf16* __restrict__ out, long long n)
{
  long long i = (long long)blockIdx.x * 256 + threadIdx.x;
  if (i < n) out[i] = f2bf(in[i]);
}

// ---------------- split qkv into head-major q,k,v (q scaled) ---------------
__global__ void split_qkv(const float* __restrict__ qkv, bf16* __restrict__ q,
                          bf16* __restrict__ k, bf16* __restrict__ v,
                          float* __restrict__ vf)
{
  long long i = (long long)blockIdx.x * 256 + threadIdx.x;
  const long long total = ROWS * (long long)INNER;
  if (i >= total) return;
  const int d  = (int)(i & 63);
  const int hh = (int)((i >> 6) & 7);
  const long long rn = i >> 9;            // b*8192+n
  const long long b  = rn >> 13;
  const long long n  = rn & (N_SEQ - 1);
  const float* p = qkv + rn * (3 * INNER) + hh * DH + d;
  const long long o = (((b * HEADS + hh) * N_SEQ) + n) * DH + d;
  q[o] = f2bf(p[0] * 0.125f);             // 64^-0.5
  k[o] = f2bf(p[INNER]);
  const float vvv = p[2 * INNER];
  v[o] = f2bf(vvv);
  vf[o] = vvv;
}

// ---------------- landmark means (from fp32 qkv for precision) -------------
__global__ void landmark_kernel(const float* __restrict__ qkv,
                                bf16* __restrict__ ql, bf16* __restrict__ kl)
{
  long long i = (long long)blockIdx.x * 256 + threadIdx.x;
  const long long total = (long long)BH * M_LM * DH;
  if (i >= total) return;
  const int d  = (int)(i & 63);
  const int m  = (int)((i >> 6) & (M_LM - 1));
  const int bh = (int)(i >> 14);
  const int b = bh >> 3, hh = bh & 7;
  float sq = 0.f, sk = 0.f;
  for (int t = 0; t < L_GRP; ++t) {
    const long long rn = (long long)b * N_SEQ + m * L_GRP + t;
    const float* p = qkv + rn * (3 * INNER);
    sq += p[hh * DH + d];
    sk += p[INNER + hh * DH + d];
  }
  ql[i] = f2bf(sq * (0.125f / L_GRP));
  kl[i] = f2bf(sk * (1.f / L_GRP));
}

// ---------------- row softmax: fp32 in-place + bf16 out --------------------
__global__ __launch_bounds__(256)
void softmax_rows(float* __restrict__ X, bf16* __restrict__ Y, int cols)
{
  __shared__ float red[256];
  const long long row = blockIdx.x;
  float* xr = X + row * (long long)cols;
  bf16*  yr = Y + row * (long long)cols;
  const int tid = threadIdx.x;
  float m = -3.0e38f;
  for (int i = tid; i < cols; i += 256) m = fmaxf(m, xr[i]);
  red[tid] = m; __syncthreads();
  for (int st = 128; st > 0; st >>= 1) { if (tid < st) red[tid] = fmaxf(red[tid], red[tid + st]); __syncthreads(); }
  const float rm = red[0]; __syncthreads();
  float s = 0.f;
  for (int i = tid; i < cols; i += 256) { float e = __expf(xr[i] - rm); xr[i] = e; s += e; }
  red[tid] = s; __syncthreads();
  for (int st = 128; st > 0; st >>= 1) { if (tid < st) red[tid] += red[tid + st]; __syncthreads(); }
  const float inv = 1.f / red[0];
  for (int i = tid; i < cols; i += 256) { float v = xr[i] * inv; xr[i] = v; yr[i] = f2bf(v); }
}

// ---------------- pinv init: per-matrix max |row-sum| / |col-sum| ----------
__global__ __launch_bounds__(256)
void pinv_absmax(const float* __restrict__ a, float* __restrict__ pm)
{
  __shared__ float red[256];
  const int mat = blockIdx.x;
  const float* A = a + (long long)mat * (M_LM * M_LM);
  const int tid = threadIdx.x;
  float rs = 0.f;
  for (int j = 0; j < M_LM; ++j) rs += fabsf(A[tid * M_LM + j]);
  red[tid] = rs; __syncthreads();
  for (int st = 128; st > 0; st >>= 1) { if (tid < st) red[tid] = fmaxf(red[tid], red[tid + st]); __syncthreads(); }
  const float maxcol = red[0]; __syncthreads();
  float cs = 0.f;
  for (int i = 0; i < M_LM; ++i) cs += fabsf(A[i * M_LM + tid]);
  red[tid] = cs; __syncthreads();
  for (int st = 128; st > 0; st >>= 1) { if (tid < st) red[tid] = fmaxf(red[tid], red[tid + st]); __syncthreads(); }
  if (tid == 0) { pm[mat * 2] = maxcol; pm[mat * 2 + 1] = red[0]; }
}

__global__ void pinv_scale(const float* __restrict__ pm, float* __restrict__ scl)
{
  if (threadIdx.x == 0) {
    float a = 0.f, b = 0.f;
    for (int i = 0; i < BH; ++i) { a = fmaxf(a, pm[i * 2]); b = fmaxf(b, pm[i * 2 + 1]); }
    scl[0] = 1.f / (a * b);   // GLOBAL max across the whole batch (as in ref)
  }
}

__global__ void pinv_initz(const float* __restrict__ a, const float* __restrict__ scl,
                           bf16* __restrict__ z)
{
  long long i = (long long)blockIdx.x * 256 + threadIdx.x;
  const long long total = (long long)BH * M_LM * M_LM;
  if (i >= total) return;
  const long long mat = i >> 16;
  const int p = (int)((i >> 8) & 255), q = (int)(i & 255);
  z[i] = f2bf(a[(mat << 16) + ((long long)q << 8) + p] * scl[0]);   // a^T * s
}

// ---------------- out = c*I - in (batched 256x256), bf16 out ---------------
__global__ void diag_sub(const float* __restrict__ in, bf16* __restrict__ out, float c)
{
  long long i = (long long)blockIdx.x * 256 + threadIdx.x;
  const long long total = (long long)BH * M_LM * M_LM;
  if (i >= total) return;
  const int p = (int)((i >> 8) & 255), q = (int)(i & 255);
  out[i] = f2bf(((p == q) ? c : 0.f) - in[i]);
}

// ---------------- depthwise conv residual + head merge -> bf16 -------------
__global__ void conv_merge(const float* __restrict__ out2, const float* __restrict__ vf,
                           const float* __restrict__ rw, bf16* __restrict__ y)
{
  long long i = (long long)blockIdx.x * 256 + threadIdx.x;
  const long long total = (long long)BH * N_SEQ * DH;
  if (i >= total) return;
  const int d = (int)(i & 63);
  const int n = (int)((i >> 6) & (N_SEQ - 1));
  const int bh = (int)(i >> 19);
  const int b = bh >> 3, hh = bh & 7;
  float acc = out2[i];
  const float* vch = vf + ((long long)bh * N_SEQ) * DH + d;
  #pragma unroll 1
  for (int k = 0; k < KER; ++k) {
    const int j = n + k - (KER / 2);
    if (j >= 0 && j < N_SEQ) acc += rw[hh * KER + k] * vch[(long long)j * DH];
  }
  y[((long long)(b * N_SEQ + n)) * INNER + hh * DH + d] = f2bf(acc);
}

// ---------------- host-side GEMM launcher ----------------------------------
static void launch_gemm(hipStream_t s, const bf16* A, long long sA, int lda,
                        const bf16* Bm, long long sB, int ldb, int tB,
                        float* C, bf16* D, long long sC, int ldc,
                        int M, int N, int K, int batch, float alpha,
                        const float* bias, const float* resid)
{
  dim3 g((unsigned)((N + 127) / 128), (unsigned)((M + 127) / 128), (unsigned)batch);
  wmma_gemm_bf16<<<g, 256, 0, s>>>(A, sA, lda, Bm, sB, ldb, tB, C, D, sC, ldc,
                                   M, N, K, alpha, bias, resid);
}

extern "C" void kernel_launch(void* const* d_in, const int* in_sizes, int n_in,
                              void* d_out, int out_size, void* d_ws, size_t ws_size,
                              hipStream_t stream)
{
  (void)in_sizes; (void)n_in; (void)out_size; (void)ws_size;
  const float* x    = (const float*)d_in[0];
  const float* lnw  = (const float*)d_in[1];
  const float* lnb  = (const float*)d_in[2];
  const float* wqkv = (const float*)d_in[3];
  const float* wout = (const float*)d_in[4];
  const float* bout = (const float*)d_in[5];
  const float* resw = (const float*)d_in[6];
  float* out = (float*)d_out;

  char* ws = (char*)d_ws;
  size_t off = 0;
  auto alloc = [&](size_t bytes) -> char* {
    off = (off + 255) & ~(size_t)255;
    char* p = ws + off; off += bytes; return p;
  };

  const long long mm  = (long long)M_LM * M_LM;              // 65536
  const long long nmL = (long long)N_SEQ * M_LM;             // 2097152
  bf16*  h_bf    = (bf16*) alloc((size_t)ROWS * D_MODEL * 2);
  bf16*  wqkv_bf = (bf16*) alloc((size_t)D_MODEL * 3 * INNER * 2);
  bf16*  wout_bf = (bf16*) alloc((size_t)INNER * D_MODEL * 2);
  float* qkv     = (float*)alloc((size_t)ROWS * 3 * INNER * 4);   // reused for pinv fp32 temps
  bf16*  q_bf    = (bf16*) alloc((size_t)BH * N_SEQ * DH * 2);
  bf16*  k_bf    = (bf16*) alloc((size_t)BH * N_SEQ * DH * 2);
  bf16*  v_bf    = (bf16*) alloc((size_t)BH * N_SEQ * DH * 2);
  float* v_f     = (float*)alloc((size_t)BH * N_SEQ * DH * 4);
  bf16*  ql_bf   = (bf16*) alloc((size_t)BH * M_LM * DH * 2);
  bf16*  kl_bf   = (bf16*) alloc((size_t)BH * M_LM * DH * 2);
  float* log1    = (float*)alloc((size_t)BH * nmL * 4);           // reused as tmp1_bf
  bf16*  attn1_bf= (bf16*) alloc((size_t)BH * nmL * 2);
  float* log3    = (float*)alloc((size_t)BH * nmL * 4);           // reused as y_bf
  bf16*  attn3_bf= (bf16*) alloc((size_t)BH * nmL * 2);
  float* a2      = (float*)alloc((size_t)BH * mm * 4);
  bf16*  a2_bf   = (bf16*) alloc((size_t)BH * mm * 2);
  bf16*  zA      = (bf16*) alloc((size_t)BH * mm * 2);
  bf16*  zB      = (bf16*) alloc((size_t)BH * mm * 2);
  bf16*  az_bf   = (bf16*) alloc((size_t)BH * mm * 2);
  bf16*  w_bf    = (bf16*) alloc((size_t)BH * mm * 2);
  float* pm      = (float*)alloc(2 * BH * 4);
  float* scl     = (float*)alloc(256);
  bf16*  a3v_bf  = (bf16*) alloc((size_t)BH * M_LM * DH * 2);
  float* out2    = (float*)alloc((size_t)BH * N_SEQ * DH * 4);

  // region reuse (stream-ordered, producers of old contents are long done)
  float* az_f    = qkv;                        // pinv temps live in dead qkv region
  float* t_f     = qkv + (size_t)BH * mm;
  bf16*  tmp1_bf = (bf16*)log1;                // attn1@pinv (bf16) in dead log1
  bf16*  y_bf    = (bf16*)log3;                // merged heads (bf16) in dead log3

  // 1) LayerNorm + weight casts
  ln_kernel<<<(unsigned)ROWS, 256, 0, stream>>>(x, lnw, lnb, h_bf);
  {
    long long nwq = (long long)D_MODEL * 3 * INNER;
    cast_kernel<<<(unsigned)((nwq + 255) / 256), 256, 0, stream>>>(wqkv, wqkv_bf, nwq);
    long long nwo = (long long)INNER * D_MODEL;
    cast_kernel<<<(unsigned)((nwo + 255) / 256), 256, 0, stream>>>(wout, wout_bf, nwo);
  }

  // 2) qkv = h @ w_qkv : [32768,512]x[512,1536]
  launch_gemm(stream, h_bf, 0, D_MODEL, wqkv_bf, 0, 3 * INNER, 0,
              qkv, nullptr, 0, 3 * INNER, (int)ROWS, 3 * INNER, D_MODEL, 1, 1.f,
              nullptr, nullptr);

  // 3) split + landmarks
  split_qkv<<<(unsigned)((ROWS * INNER + 255) / 256), 256, 0, stream>>>(qkv, q_bf, k_bf, v_bf, v_f);
  landmark_kernel<<<(unsigned)(((long long)BH * M_LM * DH + 255) / 256), 256, 0, stream>>>(qkv, ql_bf, kl_bf);

  // 4) attention logits (batched over 32 (b,h) pairs)
  const long long sQ = (long long)N_SEQ * DH, sL = (long long)M_LM * DH;
  launch_gemm(stream, q_bf,  sQ, DH, kl_bf, sL, DH, 1, log1, nullptr, nmL, M_LM,
              N_SEQ, M_LM, DH, BH, 1.f, nullptr, nullptr);                 // q  @ k_l^T
  launch_gemm(stream, ql_bf, sL, DH, kl_bf, sL, DH, 1, a2,   nullptr, mm,  M_LM,
              M_LM, M_LM, DH, BH, 1.f, nullptr, nullptr);                  // q_l@ k_l^T
  launch_gemm(stream, ql_bf, sL, DH, k_bf,  sQ, DH, 1, log3, nullptr, nmL, N_SEQ,
              M_LM, N_SEQ, DH, BH, 1.f, nullptr, nullptr);                 // q_l@ k^T

  softmax_rows<<<(unsigned)(BH * N_SEQ), 256, 0, stream>>>(log1, attn1_bf, M_LM);
  softmax_rows<<<(unsigned)(BH * M_LM), 256, 0, stream>>>(a2,   a2_bf,    M_LM);
  softmax_rows<<<(unsigned)(BH * M_LM), 256, 0, stream>>>(log3, attn3_bf, N_SEQ);

  // 5) Moore-Penrose pinv of attn2 (Newton-Schulz, batched 256^3 WMMA GEMMs)
  pinv_absmax<<<BH, 256, 0, stream>>>(a2, pm);
  pinv_scale<<<1, 64, 0, stream>>>(pm, scl);
  {
    unsigned g = (unsigned)((BH * mm + 255) / 256);
    pinv_initz<<<g, 256, 0, stream>>>(a2, scl, zA);
    bf16* zc = zA; bf16* zn = zB;
    for (int it = 0; it < PINV_ITERS; ++it) {
      launch_gemm(stream, a2_bf, mm, M_LM, zc, mm, M_LM, 0, az_f, az_bf, mm, M_LM,
                  M_LM, M_LM, M_LM, BH, 1.f, nullptr, nullptr);            // az = a@z
      diag_sub<<<g, 256, 0, stream>>>(az_f, w_bf, 7.f);                    // 7I - az
      launch_gemm(stream, az_bf, mm, M_LM, w_bf, mm, M_LM, 0, t_f, nullptr, mm, M_LM,
                  M_LM, M_LM, M_LM, BH, 1.f, nullptr, nullptr);            // az@(7I-az)
      diag_sub<<<g, 256, 0, stream>>>(t_f, w_bf, 15.f);                    // 15I - ..
      launch_gemm(stream, az_bf, mm, M_LM, w_bf, mm, M_LM, 0, t_f, nullptr, mm, M_LM,
                  M_LM, M_LM, M_LM, BH, 1.f, nullptr, nullptr);            // az@(15I-..)
      diag_sub<<<g, 256, 0, stream>>>(t_f, w_bf, 13.f);                    // 13I - ..
      launch_gemm(stream, zc, mm, M_LM, w_bf, mm, M_LM, 0, nullptr, zn, mm, M_LM,
                  M_LM, M_LM, M_LM, BH, 0.25f, nullptr, nullptr);          // z=0.25 z@..
      bf16* t = zc; zc = zn; zn = t;
    }
    zA = zc;   // final pinv (bf16)
  }

  // 6) attn3v = attn3 @ v : [256,8192]x[8192,64]
  launch_gemm(stream, attn3_bf, nmL, N_SEQ, v_bf, sQ, DH, 0, nullptr, a3v_bf,
              sL, DH, M_LM, DH, N_SEQ, BH, 1.f, nullptr, nullptr);
  // 7) tmp1 = attn1 @ pinv : [8192,256]x[256,256]
  launch_gemm(stream, attn1_bf, nmL, M_LM, zA, mm, M_LM, 0, nullptr, tmp1_bf,
              nmL, M_LM, N_SEQ, M_LM, M_LM, BH, 1.f, nullptr, nullptr);
  // 8) out2 = tmp1 @ attn3v : [8192,256]x[256,64]
  launch_gemm(stream, tmp1_bf, nmL, M_LM, a3v_bf, sL, DH, 0, out2, nullptr,
              sQ, DH, N_SEQ, DH, M_LM, BH, 1.f, nullptr, nullptr);

  // 9) depthwise conv residual + merge heads -> bf16 [32768, 512]
  conv_merge<<<(unsigned)(((long long)BH * N_SEQ * DH + 255) / 256), 256, 0, stream>>>(
      out2, v_f, resw, y_bf);

  // 10) final: out = x + y @ w_out + b_out
  launch_gemm(stream, y_bf, 0, INNER, wout_bf, 0, D_MODEL, 0, out, nullptr,
              0, D_MODEL, (int)ROWS, D_MODEL, INNER, 1, 1.f, bout, x);
}
